// GCNNet1_42812234006620
// MI455X (gfx1250) — matrix operand order repaired and checked by reading, not compiled, not run on hardware
//
#include <hip/hip_runtime.h>
#include <hip/hip_bf16.h>

// ---------------------------------------------------------------------------
// GCN (4-layer GraphConv + graph-norm + BN + ReLU + residual, graph mean pool)
// MI455X / gfx1250: wave32, WMMA f32_16x16x32_f16 for the per-layer GEMM.
// ---------------------------------------------------------------------------

#define NN 50000
#define EE 640000
#define DD 128
#define GG 256
#define LL 4
#define BN_EPS 1e-5f

typedef __attribute__((ext_vector_type(16))) _Float16 v16h;
typedef __attribute__((ext_vector_type(8)))  _Float16 v8h;
typedef __attribute__((ext_vector_type(8)))  float    v8f;

// ---------------- generic helpers ----------------

__global__ void fill_zero_kernel(float* __restrict__ p, long n) {
    long i = (long)blockIdx.x * blockDim.x + threadIdx.x;
    long stride = (long)gridDim.x * blockDim.x;
    for (; i < n; i += stride) p[i] = 0.0f;
}

__global__ void copy_kernel(float* __restrict__ dst, const float* __restrict__ src, long n) {
    long i = (long)blockIdx.x * blockDim.x + threadIdx.x;
    long stride = (long)gridDim.x * blockDim.x;
    for (; i < n; i += stride) dst[i] = src[i];
}

// ---------------- degree / norms ----------------

__global__ void degree_kernel(const int* __restrict__ esrc, const int* __restrict__ edst,
                              float* __restrict__ deg_out, float* __restrict__ deg_in) {
    int e = blockIdx.x * blockDim.x + threadIdx.x;
    if (e >= EE) return;
    atomicAdd(&deg_out[esrc[e]], 1.0f);
    atomicAdd(&deg_in[edst[e]], 1.0f);
}

// in-place: deg -> rsqrt(max(deg,1))
__global__ void norm_kernel(float* __restrict__ deg_out, float* __restrict__ deg_in) {
    int n = blockIdx.x * blockDim.x + threadIdx.x;
    if (n >= NN) return;
    deg_out[n] = rsqrtf(fmaxf(deg_out[n], 1.0f));
    deg_in[n]  = rsqrtf(fmaxf(deg_in[n], 1.0f));
}

// ---------------- edge gather-scatter:  agg[dst] += h[src] * norm_src[src] --

__global__ void edge_scatter_kernel(const float* __restrict__ h,
                                    const float* __restrict__ norm_src,
                                    const int* __restrict__ esrc,
                                    const int* __restrict__ edst,
                                    float* __restrict__ agg) {
    long t = (long)blockIdx.x * blockDim.x + threadIdx.x;   // E * (D/4) threads
    if (t >= (long)EE * (DD / 4)) return;
    int e  = (int)(t >> 5);          // D/4 == 32 chunks per edge
    int c4 = ((int)t & 31) * 4;
    int s = esrc[e], d = edst[e];
    float ns = norm_src[s];
    const float* hs = &h[(long)s * DD + c4];
    float* ad = &agg[(long)d * DD + c4];
    atomicAdd(&ad[0], hs[0] * ns);
    atomicAdd(&ad[1], hs[1] * ns);
    atomicAdd(&ad[2], hs[2] * ns);
    atomicAdd(&ad[3], hs[3] * ns);
}

// ---------------- WMMA GEMM: hpre = (agg .* norm_dst) @ W + b, then * snorm --
// In-place over `data` (block stages its 128-row tile in LDS before writing).

#define LDS_PITCH 136   // halves per row; 272B rows -> 16B aligned, spreads banks

__global__ __launch_bounds__(256) void gcn_gemm_kernel(
        float* __restrict__ data,            // [N,128]: read agg, write hpre
        const float* __restrict__ norm_dst,  // [N]
        const float* __restrict__ W,         // [128,128] row-major (k, n)
        const float* __restrict__ bias,      // [128]
        const float* __restrict__ snorm) {   // [N]
    __shared__ _Float16 As[DD * LDS_PITCH]; // A tile: [row][k]
    __shared__ _Float16 Bs[DD * LDS_PITCH]; // W^T tile: [n][k]

    const int block_row = blockIdx.x * DD;
    const int tid = threadIdx.x;

    // Stage A (convert f32->f16, fold norm_dst); guard rows >= N with zeros.
    for (int idx = tid; idx < DD * DD; idx += 256) {
        int r = idx >> 7, k = idx & 127;
        int grow = block_row + r;
        float v = 0.0f;
        if (grow < NN) v = data[(long)grow * DD + k] * norm_dst[grow];
        As[r * LDS_PITCH + k] = (_Float16)v;
    }
    // Stage W transposed: Bs[n][k] = W[k][n]; coalesced global reads.
    for (int idx = tid; idx < DD * DD; idx += 256) {
        int k = idx >> 7, n = idx & 127;
        Bs[n * LDS_PITCH + k] = (_Float16)W[idx];
    }
    __syncthreads();

    const int wave = tid >> 5;        // 0..7 -> 16-row stripe
    const int lane = tid & 31;
    const bool hi  = lane >= 16;
    const int  ml  = lane & 15;       // row (A) / col (B) within 16
    const int  m0  = wave * 16;

    v8f acc[8] = {};

    // A frag K offsets per ISA: lane<16 -> {0..7, 16..23}; lane>=16 -> {8..15, 24..31}
    const int kA0 = hi ? 8 : 0;
    const int kA1 = hi ? 24 : 16;
    // B frag K offsets (match A): lane<16 -> {0..7, 16..23}; lane>=16 -> {8..15, 24..31}
    const int kB0 = hi ? 8 : 0;

#pragma unroll
    for (int kt = 0; kt < 4; ++kt) {
        const int kbase = kt * 32;
        const _Float16* arow = &As[(m0 + ml) * LDS_PITCH + kbase];
        v8h alo = *(const v8h*)(arow + kA0);
        v8h ahi = *(const v8h*)(arow + kA1);
        v16h a;
#pragma unroll
        for (int j = 0; j < 8; ++j) { a[j] = alo[j]; a[8 + j] = ahi[j]; }

#pragma unroll
        for (int n = 0; n < 8; ++n) {
            const _Float16* brow = &Bs[(n * 16 + ml) * LDS_PITCH + kbase + kB0];
            v8h blo = *(const v8h*)(brow);
            v8h bhi = *(const v8h*)(brow + 16);
            v16h b;
#pragma unroll
            for (int j = 0; j < 8; ++j) { b[j] = blo[j]; b[8 + j] = bhi[j]; }
            acc[n] = __builtin_amdgcn_wmma_f32_16x16x32_f16(
                false, a, false, b, (short)0, acc[n], false, false);
        }
    }

    __syncthreads();   // all LDS reads done before anyone could reuse

    // Epilogue: D layout -> VGPR r: row = m0 + r + (hi?8:0), col = n*16 + ml.
#pragma unroll
    for (int n = 0; n < 8; ++n) {
        const int col = n * 16 + ml;
        const float bn = bias[col];
#pragma unroll
        for (int r = 0; r < 8; ++r) {
            int row = block_row + m0 + r + (hi ? 8 : 0);
            if (row < NN) {
                data[(long)row * DD + col] = (acc[n][r] + bn) * snorm[row];
            }
        }
    }
}

// ---------------- batch-norm stats (per-channel sum / sumsq) ----------------

__global__ void bn_stats_kernel(const float* __restrict__ h, float* __restrict__ sums) {
    int c = threadIdx.x;              // 128 threads == channels
    float s = 0.0f, sq = 0.0f;
    for (int row = blockIdx.x; row < NN; row += gridDim.x) {
        float v = h[(long)row * DD + c];
        s += v; sq += v * v;
    }
    atomicAdd(&sums[c], s);
    atomicAdd(&sums[DD + c], sq);
}

__global__ void bn_finalize_kernel(const float* __restrict__ sums,
                                   const float* __restrict__ gamma,
                                   const float* __restrict__ beta,
                                   float* __restrict__ ss) {  // [scale(128), shift(128)]
    int c = threadIdx.x;
    float invN = 1.0f / (float)NN;
    float mu  = sums[c] * invN;
    float var = fmaxf(sums[DD + c] * invN - mu * mu, 0.0f);
    float sc  = gamma[c] * rsqrtf(var + BN_EPS);
    ss[c]      = sc;
    ss[DD + c] = beta[c] - mu * sc;
}

// ---------------- BN apply + ReLU + residual (in place on h) ----------------

__global__ void bn_apply_kernel(float* __restrict__ h, const float* __restrict__ hpre,
                                const float* __restrict__ ss) {
    long i = (long)blockIdx.x * blockDim.x + threadIdx.x;
    long n = (long)NN * DD;
    long stride = (long)gridDim.x * blockDim.x;
    for (; i < n; i += stride) {
        int c = (int)(i & 127);
        float t = hpre[i] * ss[c] + ss[DD + c];
        h[i] += (t > 0.0f) ? t : 0.0f;
    }
}

// ---------------- graph mean pooling ----------------

__global__ void graph_scatter_kernel(const float* __restrict__ h,
                                     const int* __restrict__ gids,
                                     float* __restrict__ gsum,
                                     float* __restrict__ gcount) {
    long t = (long)blockIdx.x * blockDim.x + threadIdx.x;   // N * (D/4)
    if (t >= (long)NN * (DD / 4)) return;
    int node = (int)(t >> 5);
    int c4   = ((int)t & 31) * 4;
    int g = gids[node];
    const float* hn = &h[(long)node * DD + c4];
    float* gs = &gsum[(long)g * DD + c4];
    atomicAdd(&gs[0], hn[0]);
    atomicAdd(&gs[1], hn[1]);
    atomicAdd(&gs[2], hn[2]);
    atomicAdd(&gs[3], hn[3]);
    if ((t & 31) == 0) atomicAdd(&gcount[g], 1.0f);
}

__global__ void graph_divide_kernel(const float* __restrict__ gsum,
                                    const float* __restrict__ gcount,
                                    float* __restrict__ out) {
    int i = blockIdx.x * blockDim.x + threadIdx.x;
    if (i >= GG * DD) return;
    out[i] = gsum[i] / fmaxf(gcount[i >> 7], 1.0f);
}

// ---------------------------------------------------------------------------

static inline int cdiv(long a, int b) { return (int)((a + b - 1) / b); }

extern "C" void kernel_launch(void* const* d_in, const int* in_sizes, int n_in,
                              void* d_out, int out_size, void* d_ws, size_t ws_size,
                              hipStream_t stream) {
    (void)in_sizes; (void)n_in; (void)out_size; (void)ws_size;
    // setup_inputs() order:
    const float* nodes_feat = (const float*)d_in[0];   // [N,D]
    const float* snorm      = (const float*)d_in[1];   // [N,1]
    const float* Ws         = (const float*)d_in[2];   // [L,D,D]
    const float* bs         = (const float*)d_in[3];   // [L,D]
    const float* gammas     = (const float*)d_in[4];   // [L,D]
    const float* betas      = (const float*)d_in[5];   // [L,D]
    const int*   edge_src   = (const int*)d_in[6];     // [E]
    const int*   edge_dst   = (const int*)d_in[7];     // [E]
    const int*   graph_ids  = (const int*)d_in[8];     // [N]
    float* out = (float*)d_out;                        // [G,D]

    // Workspace layout (floats)
    float* ws = (float*)d_ws;
    float* h_cur    = ws;                       ws += (long)NN * DD;  // 6.4M
    float* agg      = ws;                       ws += (long)NN * DD;  // doubles as hpre
    float* norm_src = ws;                       ws += NN;             // deg_out -> norm
    float* norm_dst = ws;                       ws += NN;             // deg_in  -> norm
    float* bnsums   = ws;                       ws += 2 * DD;
    float* bnss     = ws;                       ws += 2 * DD;
    float* gsum     = ws;                       ws += (long)GG * DD;
    float* gcount   = ws;                       ws += GG;

    const long ND = (long)NN * DD;

    // degrees -> norms
    fill_zero_kernel<<<cdiv(2L * NN, 256), 256, 0, stream>>>(norm_src, NN);
    fill_zero_kernel<<<cdiv(2L * NN, 256), 256, 0, stream>>>(norm_dst, NN);
    degree_kernel<<<cdiv(EE, 256), 256, 0, stream>>>(edge_src, edge_dst, norm_src, norm_dst);
    norm_kernel<<<cdiv(NN, 256), 256, 0, stream>>>(norm_src, norm_dst);

    // h = nodes_feat
    copy_kernel<<<2048, 256, 0, stream>>>(h_cur, nodes_feat, ND);

    const int gemm_blocks = cdiv(NN, 128);   // 391
    for (int l = 0; l < LL; ++l) {
        const float* Wl = Ws + (long)l * DD * DD;
        const float* bl = bs + (long)l * DD;
        const float* gl = gammas + (long)l * DD;
        const float* tl = betas + (long)l * DD;

        fill_zero_kernel<<<4096, 256, 0, stream>>>(agg, ND);
        fill_zero_kernel<<<1, 256, 0, stream>>>(bnsums, 2 * DD);

        edge_scatter_kernel<<<cdiv((long)EE * (DD / 4), 256), 256, 0, stream>>>(
            h_cur, norm_src, edge_src, edge_dst, agg);

        gcn_gemm_kernel<<<gemm_blocks, 256, 0, stream>>>(agg, norm_dst, Wl, bl, snorm);

        bn_stats_kernel<<<512, 128, 0, stream>>>(agg, bnsums);
        bn_finalize_kernel<<<1, 128, 0, stream>>>(bnsums, gl, tl, bnss);
        bn_apply_kernel<<<4096, 256, 0, stream>>>(h_cur, agg, bnss);
    }

    // graph mean pooling
    fill_zero_kernel<<<cdiv((long)GG * DD + GG, 256), 256, 0, stream>>>(gsum, (long)GG * DD);
    fill_zero_kernel<<<1, 256, 0, stream>>>(gcount, GG);
    graph_scatter_kernel<<<cdiv((long)NN * (DD / 4), 256), 256, 0, stream>>>(
        h_cur, graph_ids, gsum, gcount);
    graph_divide_kernel<<<cdiv((long)GG * DD, 256), 256, 0, stream>>>(gsum, gcount, out);
}
